// DeformableConvolutionLayer_3736621547927
// MI455X (gfx1250) — compile-verified, matrix operand order checked by reading, not run on hardware
//
#include <hip/hip_runtime.h>

typedef __attribute__((ext_vector_type(16))) __bf16 v16bf;
typedef __attribute__((ext_vector_type(8)))  float  v8f;

#define B_   16
#define C_   256
#define O_   256
#define H_   56
#define W_   56
#define HW_  3136
#define MT_  128          // output rows per workgroup
#define NWGS 392          // 50176 / 128
#define FRAG_USH 512      // 32 lanes * 16 ushorts per B fragment
#define W2_USH  (9 * 8 * 16 * FRAG_USH)   // taps * kchunks * ntiles * frag
#define SBUF_USH (8 * 8 * 32 * 16)        // one sampled A tile: 64 KB

static __device__ __forceinline__ unsigned short f2bf(float f) {
  unsigned u = __float_as_uint(f);
  unsigned r = (u + 0x7FFFu + ((u >> 16) & 1u)) >> 16;   // round-to-nearest-even
  return (unsigned short)r;
}

// WMMA 16-bit fragment interleave (CDNA5 ISA, 16-bit A 16x32 table; B mirrored):
// element (kk in 0..31, j in 0..15) -> lane = j + 16*((kk>>3)&1), e = (kk&7) + 8*(kk>>4)
static __device__ __forceinline__ int frag_lane(int kk, int j) { return j + (((kk >> 3) & 1) << 4); }
static __device__ __forceinline__ int frag_elem(int kk)        { return (kk & 7) + (((kk >> 4) & 1) << 3); }

// ---------------------------------------------------------------------------
// Kernel 1: repack weight[O][C][3][3] fp32 -> bf16 B fragments in workspace
// ---------------------------------------------------------------------------
__global__ __launch_bounds__(256) void repack_kernel(const float* __restrict__ w,
                                                     unsigned short* __restrict__ W2) {
  int t = blockIdx.x * 256 + threadIdx.x;
  if (t >= 9 * C_ * O_) return;
  int tap = t % 9;
  int ci  = (t / 9) % C_;
  int o   = t / (9 * C_);
  float v = w[(size_t)(o * C_ + ci) * 9 + tap];
  int kchunk = ci >> 5, kk = ci & 31;
  int ntile  = o >> 4,  n  = o & 15;
  int frag = (tap * 8 + kchunk) * 16 + ntile;
  W2[(size_t)frag * FRAG_USH + frag_lane(kk, n) * 16 + frag_elem(kk)] = f2bf(v);
}

// ---------------------------------------------------------------------------
// Kernel 2: fused bilinear-sample (im2col in LDS, bf16) + WMMA GEMM,
//           double-buffered so tap t+1's gathers overlap tap t's WMMAs.
// ---------------------------------------------------------------------------
struct SampCtx {
  const float* xb;     // x base for this row's batch
  const float* offp;   // offset base for this row's (b, y, x)
  int oy, ox;          // output spatial position
  int tile, m;         // A-tile / row-in-tile of this thread's samples
  int cBeg;            // first input channel this thread samples
};

static __device__ __forceinline__ void sample_tap(const SampCtx& c, int tap,
                                                  unsigned short* __restrict__ S) {
  const int kh = tap / 3, kw = tap - kh * 3;
  const float py  = (float)(c.oy + kh - 1) + c.offp[(size_t)(2 * tap) * HW_];
  const float px  = (float)(c.ox + kw - 1) + c.offp[(size_t)(2 * tap + 1) * HW_];
  const float y0f = floorf(py), x0f = floorf(px);
  const float wy1 = py - y0f, wx1 = px - x0f;
  const float wy0 = 1.f - wy1, wx0 = 1.f - wx1;
  const int y0 = (int)y0f, x0 = (int)x0f, y1 = y0 + 1, x1 = x0 + 1;
  const float vy0 = (y0 >= 0 && y0 < H_) ? 1.f : 0.f;
  const float vy1 = (y1 >= 0 && y1 < H_) ? 1.f : 0.f;
  const float vx0 = (x0 >= 0 && x0 < W_) ? 1.f : 0.f;
  const float vx1 = (x1 >= 0 && x1 < W_) ? 1.f : 0.f;
  const int y0c = min(max(y0, 0), H_ - 1), y1c = min(max(y1, 0), H_ - 1);
  const int x0c = min(max(x0, 0), W_ - 1), x1c = min(max(x1, 0), W_ - 1);
  const int i00 = y0c * W_ + x0c, i01 = y0c * W_ + x1c;
  const int i10 = y1c * W_ + x0c, i11 = y1c * W_ + x1c;
  const float w00 = wy0 * wx0 * vy0 * vx0;
  const float w01 = wy0 * wx1 * vy0 * vx1;
  const float w10 = wy1 * wx0 * vy1 * vx0;
  const float w11 = wy1 * wx1 * vy1 * vx1;
  for (int cc = 0; cc < 128; ++cc) {
    const int ch = c.cBeg + cc;
    const float* __restrict__ xc = c.xb + (size_t)ch * HW_;
    const float v = w00 * xc[i00] + w01 * xc[i01] + w10 * xc[i10] + w11 * xc[i11];
    const int kk = ch & 31, kchunk = ch >> 5;
    S[(((c.tile * 8 + kchunk) * 32 + frag_lane(kk, c.m)) << 4) + frag_elem(kk)] = f2bf(v);
  }
}

__global__ __launch_bounds__(256) void dcn_gemm(const float* __restrict__ xin,
                                                const float* __restrict__ off,
                                                const unsigned short* __restrict__ W2,
                                                float* __restrict__ out) {
  // Two sampled A tiles in WMMA A-fragment layout: 2 x 64 KB dynamic LDS.
  extern __shared__ unsigned short Sdyn[];
  unsigned short* const Sb0 = Sdyn;
  unsigned short* const Sb1 = Sdyn + SBUF_USH;

  const int tid  = threadIdx.x;
  const int wave = tid >> 5;
  const int lane = tid & 31;
  const int wg   = blockIdx.x;

  v8f acc[16];
  const v8f vzero = {0.f, 0.f, 0.f, 0.f, 0.f, 0.f, 0.f, 0.f};
#pragma unroll
  for (int i = 0; i < 16; ++i) acc[i] = vzero;

  // Sampling assignment: 2 threads per tile-row, 128 channels each
  SampCtx ctx;
  {
    const int p   = tid >> 1;
    const int g   = wg * MT_ + p;
    const int b   = g / HW_;
    const int rem = g - b * HW_;
    ctx.oy   = rem / W_;
    ctx.ox   = rem - ctx.oy * W_;
    ctx.xb   = xin + (size_t)b * C_ * HW_;
    ctx.offp = off + (size_t)b * 18 * HW_ + ctx.oy * W_ + ctx.ox;
    ctx.tile = p >> 4;
    ctx.m    = p & 15;
    ctx.cBeg = (tid & 1) * 128;
  }

  // Prologue: fill buffer 0 with tap 0.
  sample_tap(ctx, 0, Sb0);
  __syncthreads();

  for (int tap = 0; tap < 9; ++tap) {
    unsigned short* const Scur = (tap & 1) ? Sb1 : Sb0;
    unsigned short* const Snxt = (tap & 1) ? Sb0 : Sb1;

    // Issue next tap's gathers first: their latency hides under this tap's WMMAs.
    if (tap < 8) sample_tap(ctx, tap + 1, Snxt);

    const unsigned short* __restrict__ Wt = W2 + (size_t)tap * 8 * 16 * FRAG_USH;
    if (tap < 8)  // warm the next tap's weights (global_prefetch_b8)
      __builtin_prefetch(Wt + (size_t)8 * 16 * FRAG_USH + lane * 16, 0, 1);

#pragma unroll
    for (int kc = 0; kc < 8; ++kc) {
      const v16bf a =
          *reinterpret_cast<const v16bf*>(&Scur[(((wave * 8 + kc) * 32 + lane) << 4)]);
      const unsigned short* __restrict__ Wk = Wt + (size_t)kc * 16 * FRAG_USH + lane * 16;
#pragma unroll
      for (int nt = 0; nt < 16; ++nt) {
        const v16bf bf = *reinterpret_cast<const v16bf*>(Wk + (size_t)nt * FRAG_USH);
        acc[nt] = __builtin_amdgcn_wmma_f32_16x16x32_bf16(
            false, a, false, bf, (short)0, acc[nt], false, false);
      }
    }
    // One barrier per tap: makes Snxt writes visible for the next iteration's
    // WMMAs and protects Scur from being overwritten while still being read.
    __syncthreads();
  }

  // ---- store conv result (pre-BN) ----
  // C/D layout: lane(0-15): N=lane, VGPR r -> M=r;  lane(16-31): M=r+8
  const int nloc = lane & 15;
  const int mOff = (lane >> 4) * 8;
  int obase[8];
#pragma unroll
  for (int r = 0; r < 8; ++r) {
    const int gm = wg * MT_ + wave * 16 + mOff + r;
    const int bb = gm / HW_;
    const int rr = gm - bb * HW_;
    obase[r] = bb * O_ * HW_ + rr;
  }
#pragma unroll
  for (int nt = 0; nt < 16; ++nt) {
    const int n = nt * 16 + nloc;
#pragma unroll
    for (int r = 0; r < 8; ++r) {
      out[(size_t)obase[r] + (size_t)n * HW_] = acc[nt][r];
    }
  }
}

// ---------------------------------------------------------------------------
// Kernel 3: per-channel mean / rstd (training BN, biased variance)
// ---------------------------------------------------------------------------
__global__ __launch_bounds__(256) void stats_kernel(const float* __restrict__ y,
                                                    float* __restrict__ stats) {
  const int o = blockIdx.x;
  float s = 0.f, sq = 0.f;
  for (int b = 0; b < B_; ++b) {
    const float* __restrict__ p = y + (size_t)(b * O_ + o) * HW_;
    for (int j = threadIdx.x; j < HW_; j += 256) {
      const float v = p[j];
      s += v;
      sq += v * v;
    }
  }
  __shared__ float rs[256], rq[256];
  rs[threadIdx.x] = s;
  rq[threadIdx.x] = sq;
  __syncthreads();
  for (int st = 128; st > 0; st >>= 1) {
    if (threadIdx.x < st) {
      rs[threadIdx.x] += rs[threadIdx.x + st];
      rq[threadIdx.x] += rq[threadIdx.x + st];
    }
    __syncthreads();
  }
  if (threadIdx.x == 0) {
    const float N = (float)(B_ * HW_);
    const float mean = rs[0] / N;
    const float var  = rq[0] / N - mean * mean;
    stats[2 * o]     = mean;
    stats[2 * o + 1] = rsqrtf(var + 1e-5f);
  }
}

// ---------------------------------------------------------------------------
// Kernel 4: normalize + scale/shift + ReLU, in place
// ---------------------------------------------------------------------------
__global__ __launch_bounds__(256) void bn_relu_kernel(float* __restrict__ y,
                                                      const float* __restrict__ stats,
                                                      const float* __restrict__ gamma,
                                                      const float* __restrict__ beta) {
  const size_t idx = (size_t)blockIdx.x * 256 + threadIdx.x;
  const int o = (int)((idx / HW_) & (O_ - 1));
  float v = y[idx];
  v = (v - stats[2 * o]) * stats[2 * o + 1] * gamma[o] + beta[o];
  y[idx] = v > 0.f ? v : 0.f;
}

// ---------------------------------------------------------------------------
extern "C" void kernel_launch(void* const* d_in, const int* in_sizes, int n_in,
                              void* d_out, int out_size, void* d_ws, size_t ws_size,
                              hipStream_t stream) {
  const float* x      = (const float*)d_in[0];
  const float* offset = (const float*)d_in[1];
  const float* weight = (const float*)d_in[2];
  const float* gamma  = (const float*)d_in[3];
  const float* beta   = (const float*)d_in[4];
  float* out = (float*)d_out;

  unsigned short* W2 = (unsigned short*)d_ws;
  float* stats = (float*)((char*)d_ws + (size_t)W2_USH * sizeof(unsigned short));

  repack_kernel<<<(9 * C_ * O_ + 255) / 256, 256, 0, stream>>>(weight, W2);
  dcn_gemm<<<NWGS, 256, 2 * SBUF_USH * sizeof(unsigned short), stream>>>(x, offset, W2, out);
  stats_kernel<<<O_, 256, 0, stream>>>(out, stats);
  bn_relu_kernel<<<(B_ * O_ * HW_) / 256, 256, 0, stream>>>(out, stats, gamma, beta);
}
